// PrimitiveTokenizer_66949950210383
// MI455X (gfx1250) — matrix operand order repaired and checked by reading, not compiled, not run on hardware
//
#include <hip/hip_runtime.h>
#include <hip/hip_bf16.h>
#include <math.h>

// ---------------------------------------------------------------------------
// PrimitiveTokenizer for MI455X (gfx1250, wave32, WMMA f32_16x16x32_f16)
// ---------------------------------------------------------------------------

typedef _Float16 h16;
typedef __attribute__((ext_vector_type(16))) _Float16 v16h;
typedef __attribute__((ext_vector_type(8)))  _Float16 v8h;
typedef __attribute__((ext_vector_type(8)))  float    v8f;

#define NROW 4096
#define SS   64
#define DD   256
#define KK   8

// f16 workspace layout (offsets in half-elements)
#define OFF_W1T  0u          // [8][256][64]   w1^T, K padded 48->64 with zeros
#define OFF_W2T  131072u     // [8][256][256]  w2^T
#define OFF_FW1T 655360u     // [256][768]     fuse_w1^T
#define OFF_FW2T 851968u     // [256][256]     fuse_w2^T
#define TOT_W    917504u
#define OFF_HCAT 917504u     // [4096][768]    [h_num | type_emb | layer_emb] f16

__device__ __forceinline__ float gelu_exact(float x) {
  return 0.5f * x * (1.0f + erff(x * 0.70710678118654752f));
}

// A fragment (16x32 f16) in this code's contiguous k order (not the ISA table order):
//   lanes 0-15 and 16-31 both hold row m = lane%16
//   K = half*16 + [0,16) (contiguous, matching B fragment)
__device__ __forceinline__ v16h ld_a_frag(const h16* base, int m, int lda,
                                          int kofs, int half) {
  const h16* p = base + m * lda + kofs + half * 16;
  v8h lo = *(const v8h*)(p);
  v8h hi = *(const v8h*)(p + 8);
  v16h a;
#pragma unroll
  for (int i = 0; i < 8; ++i) { a[i] = lo[i]; a[i + 8] = hi[i]; }
  return a;
}

// B fragment (32x16 f16): lane holds column c = lane%16,
// K = (lane/16)*16 + [0,16) -> 16 contiguous halves of row c of W^T
__device__ __forceinline__ v16h ld_b_frag(const h16* p) {
  v8h lo = *(const v8h*)(p);
  v8h hi = *(const v8h*)(p + 8);
  v16h b;
#pragma unroll
  for (int i = 0; i < 8; ++i) { b[i] = lo[i]; b[i + 8] = hi[i]; }
  return b;
}

__device__ __forceinline__ v8f wmma16(v16h a, v16h b, v8f c) {
  return __builtin_amdgcn_wmma_f32_16x16x32_f16(
      /*neg_a=*/false, a, /*neg_b=*/false, b,
      /*c_mod=*/(short)0, c, /*reuse_a=*/false, /*reuse_b=*/false);
}

// ---------------------------------------------------------------------------
// Kernel 0: convert + transpose all weights into f16 workspace
// ---------------------------------------------------------------------------
__global__ __launch_bounds__(256) void tok_prep(
    const float* __restrict__ w1, const float* __restrict__ w2,
    const float* __restrict__ fw1, const float* __restrict__ fw2,
    h16* ws) {
  for (unsigned i = blockIdx.x * blockDim.x + threadIdx.x; i < TOT_W;
       i += gridDim.x * blockDim.x) {
    h16 o;
    if (i < OFF_W2T) {                       // W1T [8][256][64], pad 48->64
      unsigned k = i >> 14, r = i & 16383u, c = r >> 6, ii = r & 63u;
      o = (ii < 48u) ? (h16)w1[k * 12288u + ii * 256u + c] : (h16)0.f;
    } else if (i < OFF_FW1T) {               // W2T [8][256][256]
      unsigned j = i - OFF_W2T, k = j >> 16, r = j & 65535u;
      unsigned c = r >> 8, ii = r & 255u;
      o = (h16)w2[k * 65536u + ii * 256u + c];
    } else if (i < OFF_FW2T) {               // FW1T [256][768]
      unsigned j = i - OFF_FW1T, c = j / 768u, ii = j - c * 768u;
      o = (h16)fw1[ii * 256u + c];
    } else {                                 // FW2T [256][256]
      unsigned j = i - OFF_FW2T, c = j >> 8, ii = j & 255u;
      o = (h16)fw2[ii * 256u + c];
    }
    ws[i] = o;
  }
}

// ---------------------------------------------------------------------------
// Kernel 1: per-row (64 slots) Fourier -> routed MLP (WMMA) -> masked pool
// One workgroup per row n. 8 waves; wave w owns output n-tiles {2w, 2w+1}.
// ---------------------------------------------------------------------------
__global__ __launch_bounds__(256) void tok_slots(
    const float* __restrict__ values, const int* __restrict__ kinds,
    const int* __restrict__ mask, const int* __restrict__ prim_type,
    const int* __restrict__ layer_id, const float* __restrict__ B_ff,
    const float* __restrict__ b1, const float* __restrict__ b2,
    const float* __restrict__ kind_emb, const float* __restrict__ type_emb,
    const float* __restrict__ layer_emb, h16* ws) {
  __shared__ __align__(16) h16 sFF[64 * 72];    // fourier features, K-pad 64
  __shared__ __align__(16) h16 sH1[64 * 264];   // layer-1 activations (f16)
  __shared__ float sPool[256];
  __shared__ int   sKind[64];
  __shared__ float sMaskF[64];
  __shared__ float sCnt[8];
  __shared__ float sInvDen;

  const int n = blockIdx.x;
  const int tid = threadIdx.x;
  const int wid = tid >> 5, lane = tid & 31, l16 = lane & 15, half = lane >> 4;

  if (tid < 64) {
    sKind[tid] = kinds[n * 64 + tid];
    sMaskF[tid] = (float)mask[n * 64 + tid];
  }
  sPool[tid] = 0.f;

  // Fourier features: ff = [sin(2*pi*v*B), cos(2*pi*v*B)], zero-pad 48..63
  for (int idx = tid; idx < 64 * 64; idx += 256) {
    int s = idx >> 6, f = idx & 63;
    float v = values[n * 64 + s];
    float r = 0.f;
    if (f < 24)      r = sinf(6.28318530717958648f * v * B_ff[f]);
    else if (f < 48) r = cosf(6.28318530717958648f * v * B_ff[f - 24]);
    sFF[s * 72 + f] = (h16)r;
  }
  __syncthreads();

  if (tid == 0) {   // per-kind masked counts + pooling denominator
    float cnt[8];
#pragma unroll
    for (int k = 0; k < 8; ++k) cnt[k] = 0.f;
    float den = 0.f;
    for (int s = 0; s < 64; ++s) { den += sMaskF[s]; cnt[sKind[s]] += sMaskF[s]; }
#pragma unroll
    for (int k = 0; k < 8; ++k) sCnt[k] = cnt[k];
    sInvDen = 1.0f / fmaxf(den, 1.0f);
  }

  int myK[4];
#pragma unroll
  for (int mt = 0; mt < 4; ++mt) myK[mt] = sKind[mt * 16 + l16];

  const v16h zh = {};
  const v8f  zf = {};

  // ---------------- Layer 1: [64 x 64] @ [64 x 256], kind-routed -----------
  v16h aU[4][2];
#pragma unroll
  for (int mt = 0; mt < 4; ++mt)
#pragma unroll
    for (int ks = 0; ks < 2; ++ks)
      aU[mt][ks] = ld_a_frag(sFF, mt * 16 + l16, 72, ks * 32, half);

  v8f acc[4][2];
#pragma unroll
  for (int mt = 0; mt < 4; ++mt)
#pragma unroll
    for (int j = 0; j < 2; ++j) acc[mt][j] = zf;

  const h16* W1T = ws + OFF_W1T;
  for (int k = 0; k < 8; ++k) {
#pragma unroll
    for (int ks = 0; ks < 2; ++ks) {
      v16h am[4];
#pragma unroll
      for (int mt = 0; mt < 4; ++mt) am[mt] = (myK[mt] == k) ? aU[mt][ks] : zh;
#pragma unroll
      for (int j = 0; j < 2; ++j) {
        int c = (wid * 2 + j) * 16 + l16;
        v16h b = ld_b_frag(W1T + (unsigned)(k * 256 + c) * 64 + ks * 32 + half * 16);
#pragma unroll
        for (int mt = 0; mt < 4; ++mt) acc[mt][j] = wmma16(am[mt], b, acc[mt][j]);
      }
    }
  }

  // bias + exact GELU, store f16 activations to LDS
#pragma unroll
  for (int mt = 0; mt < 4; ++mt)
#pragma unroll
    for (int j = 0; j < 2; ++j) {
      int c = (wid * 2 + j) * 16 + l16;
#pragma unroll
      for (int g = 0; g < 8; ++g) {
        int m = mt * 16 + g + 8 * half;
        float x = acc[mt][j][g] + b1[sKind[m] * 256 + c];
        sH1[m * 264 + c] = (h16)gelu_exact(x);
      }
    }
  __syncthreads();

  // ---------------- Layer 2: [64 x 256] @ [256 x 256], kind-routed ---------
  v8f acc2[4][2];
#pragma unroll
  for (int mt = 0; mt < 4; ++mt)
#pragma unroll
    for (int j = 0; j < 2; ++j) acc2[mt][j] = zf;

  const h16* W2T = ws + OFF_W2T;
  for (int k = 0; k < 8; ++k) {
    for (int ks = 0; ks < 8; ++ks) {
      v16h am[4];
#pragma unroll
      for (int mt = 0; mt < 4; ++mt) {
        v16h a = ld_a_frag(sH1, mt * 16 + l16, 264, ks * 32, half);
        am[mt] = (myK[mt] == k) ? a : zh;
      }
#pragma unroll
      for (int j = 0; j < 2; ++j) {
        int c = (wid * 2 + j) * 16 + l16;
        v16h b = ld_b_frag(W2T + (unsigned)(k * 256 + c) * 256 + ks * 32 + half * 16);
#pragma unroll
        for (int mt = 0; mt < 4; ++mt) acc2[mt][j] = wmma16(am[mt], b, acc2[mt][j]);
      }
    }
  }

  // ---------------- masked mean pooling (slot activations never hit HBM) ---
#pragma unroll
  for (int mt = 0; mt < 4; ++mt)
#pragma unroll
    for (int j = 0; j < 2; ++j) {
      int c = (wid * 2 + j) * 16 + l16;
      float part = 0.f;
#pragma unroll
      for (int g = 0; g < 8; ++g) {
        int m = mt * 16 + g + 8 * half;
        part += sMaskF[m] * acc2[mt][j][g];
      }
      atomicAdd(&sPool[c], part);   // ds_add_f32
    }
  __syncthreads();

  // epilogue: fold masked (b2 + kind_emb) via per-kind counts; write HCAT f16
  {
    int c = tid;  // 256 threads == 256 channels
    float extra = 0.f;
#pragma unroll
    for (int k = 0; k < 8; ++k)
      extra += sCnt[k] * (b2[k * 256 + c] + kind_emb[k * 256 + c]);
    float hn = (sPool[c] + extra) * sInvDen;
    h16* HC = ws + OFF_HCAT + (unsigned)n * 768;
    HC[c] = (h16)hn;
    int pt = prim_type[n], lid = layer_id[n];
    HC[256 + c] = (h16)type_emb[pt * 256 + c];
    HC[512 + c] = (h16)layer_emb[(unsigned)lid * 256 + c];
  }
}

// ---------------------------------------------------------------------------
// Kernel 2: fuse MLP (768 -> 256 -> 256) + meta linear. 16 rows per block.
// ---------------------------------------------------------------------------
__global__ __launch_bounds__(256) void tok_fuse(
    const h16* ws, const float* __restrict__ fb1, const float* __restrict__ fb2,
    const float* __restrict__ meta, const float* __restrict__ meta_w,
    const float* __restrict__ meta_b, float* __restrict__ out) {
  __shared__ __align__(16) h16 sA[16 * 776];
  __shared__ __align__(16) h16 sH[16 * 264];

  const int r0 = blockIdx.x * 16;
  const int tid = threadIdx.x;
  const int wid = tid >> 5, lane = tid & 31, l16 = lane & 15, half = lane >> 4;
  const v8f zf = {};

  const h16* HC = ws + OFF_HCAT;
  for (int t = tid; t < 16 * 96; t += 256) {   // stage 16 x 768 halves
    int r = t / 96, ch = t % 96;
    *(v8h*)(sA + r * 776 + ch * 8) =
        *(const v8h*)(HC + (unsigned)(r0 + r) * 768 + ch * 8);
  }
  __syncthreads();

  // fuse layer 1: [16 x 768] @ [768 x 256]
  v8f acc[2] = {zf, zf};
  const h16* FW1T = ws + OFF_FW1T;
  for (int ks = 0; ks < 24; ++ks) {
    v16h a = ld_a_frag(sA, l16, 776, ks * 32, half);
#pragma unroll
    for (int j = 0; j < 2; ++j) {
      int c = (wid * 2 + j) * 16 + l16;
      v16h b = ld_b_frag(FW1T + (unsigned)c * 768 + ks * 32 + half * 16);
      acc[j] = wmma16(a, b, acc[j]);
    }
  }
#pragma unroll
  for (int j = 0; j < 2; ++j) {
    int c = (wid * 2 + j) * 16 + l16;
#pragma unroll
    for (int g = 0; g < 8; ++g) {
      int m = g + 8 * half;
      sH[m * 264 + c] = (h16)gelu_exact(acc[j][g] + fb1[c]);
    }
  }
  __syncthreads();

  // fuse layer 2: [16 x 256] @ [256 x 256]
  v8f acc2[2] = {zf, zf};
  const h16* FW2T = ws + OFF_FW2T;
  for (int ks = 0; ks < 8; ++ks) {
    v16h a = ld_a_frag(sH, l16, 264, ks * 32, half);
#pragma unroll
    for (int j = 0; j < 2; ++j) {
      int c = (wid * 2 + j) * 16 + l16;
      v16h b = ld_b_frag(FW2T + (unsigned)c * 256 + ks * 32 + half * 16);
      acc2[j] = wmma16(a, b, acc2[j]);
    }
  }

  // epilogue: + fuse_b2 + meta @ meta_w + meta_b, f32 out
#pragma unroll
  for (int j = 0; j < 2; ++j) {
    int c = (wid * 2 + j) * 16 + l16;
#pragma unroll
    for (int g = 0; g < 8; ++g) {
      int row = r0 + g + 8 * half;
      float x = acc2[j][g] + fb2[c] + meta_b[c];
#pragma unroll
      for (int q = 0; q < 4; ++q) x += meta[row * 4 + q] * meta_w[q * 256 + c];
      out[(unsigned)row * 256 + c] = x;
    }
  }
}

// ---------------------------------------------------------------------------
extern "C" void kernel_launch(void* const* d_in, const int* in_sizes, int n_in,
                              void* d_out, int out_size, void* d_ws,
                              size_t ws_size, hipStream_t stream) {
  const float* values    = (const float*)d_in[0];
  const int*   kinds     = (const int*)d_in[1];
  const int*   mask      = (const int*)d_in[2];
  const int*   prim_type = (const int*)d_in[3];
  const int*   layer_id  = (const int*)d_in[4];
  const float* meta      = (const float*)d_in[5];
  const float* B_ff      = (const float*)d_in[6];
  const float* kind_emb  = (const float*)d_in[7];
  const float* type_emb  = (const float*)d_in[8];
  const float* layer_emb = (const float*)d_in[9];
  const float* w1        = (const float*)d_in[10];
  const float* b1        = (const float*)d_in[11];
  const float* w2        = (const float*)d_in[12];
  const float* b2        = (const float*)d_in[13];
  const float* fw1       = (const float*)d_in[14];
  const float* fb1       = (const float*)d_in[15];
  const float* fw2       = (const float*)d_in[16];
  const float* fb2       = (const float*)d_in[17];
  const float* meta_w    = (const float*)d_in[18];
  const float* meta_b    = (const float*)d_in[19];
  h16*   ws  = (h16*)d_ws;
  float* out = (float*)d_out;

  tok_prep<<<1024, 256, 0, stream>>>(w1, w2, fw1, fw2, ws);
  tok_slots<<<NROW, 256, 0, stream>>>(values, kinds, mask, prim_type, layer_id,
                                      B_ff, b1, b2, kind_emb, type_emb,
                                      layer_emb, ws);
  tok_fuse<<<NROW / 16, 256, 0, stream>>>(ws, fb1, fb2, meta, meta_w, meta_b,
                                          out);
}
